// TFGPTNeoSelfAttention_33947421507867
// MI455X (gfx1250) — matrix operand order repaired and checked by reading, not compiled, not run on hardware
//
#include <hip/hip_runtime.h>

// ---------------------------------------------------------------------------
// GPT-Neo self-attention for MI455X (gfx1250, wave32, WMMA + TDM).
// bf16 WMMA (16x16x32, fp32 accum) everywhere; weights pre-transposed so all
// LDS staging is straight 2D tiles -> Tensor Data Mover double buffering.
// Two-pass flash attention honors the "mask with global-min" quirk without
// materializing the 536MB score tensor.
// ---------------------------------------------------------------------------

typedef unsigned short u16;
typedef unsigned int   u32;
typedef unsigned long long u64;
typedef __attribute__((ext_vector_type(4)))  unsigned int u32x4;
typedef __attribute__((ext_vector_type(8)))  int          i32x8;
typedef __attribute__((ext_vector_type(4)))  int          i32x4;
typedef __attribute__((ext_vector_type(4)))  float        f32x4;
typedef __attribute__((ext_vector_type(16))) __bf16       v16bf;
typedef __attribute__((ext_vector_type(8)))  float        v8f;

#define BB 2
#define SS 2048
#define EE 2048
#define HH 16
#define DD 128

#if defined(__HIP_DEVICE_COMPILE__) && __has_builtin(__builtin_amdgcn_tensor_load_to_lds)
#define USE_TDM 1
#else
#define USE_TDM 0
#endif

union FragU { u32x4 q[2]; v16bf v; u16 s[16]; };

__device__ __forceinline__ u16 f2bf(float f) {
  u32 u = __float_as_uint(f);
  u32 r = u + 0x7FFFu + ((u >> 16) & 1u);   // round-to-nearest-even
  return (u16)(r >> 16);
}

__device__ __forceinline__ v16bf make_frag(u32x4 a, u32x4 b) {
  FragU f; f.q[0] = a; f.q[1] = b; return f.v;
}

__device__ __forceinline__ v8f v8f_zero() {
  v8f z = {0.f, 0.f, 0.f, 0.f, 0.f, 0.f, 0.f, 0.f};
  return z;
}

__device__ __forceinline__ v8f wmma_bf16(v16bf a, v16bf b, v8f c) {
  return __builtin_amdgcn_wmma_f32_16x16x32_bf16(false, a, false, b,
                                                 (short)0, c, false, false);
}

#if USE_TDM
typedef __attribute__((address_space(3))) void lds_void_t;
__device__ __forceinline__ u32 lds_raw(void* p) {
  return (u32)(u64)(lds_void_t*)p;   // raw LDS byte offset
}

// 2D tile load via Tensor Data Mover: tile_d0 contiguous bf16 elems per row,
// tile_d1 rows, row stride strideD0 elems, packed contiguously into LDS.
// D# packing per CDNA5 ISA (data_size = 2 bytes).
__device__ __forceinline__ void tdm_load_2d(u32 ldsAddr, const void* gp,
                                            u32 tenD0, u32 tenD1,
                                            u32 tilD0, u32 tilD1, u32 strD0) {
  const u64 ga = (u64)gp;
  u32x4 g0;
  g0.x = 1u;                                      // count=1 (valid user D#)
  g0.y = ldsAddr;                                 // lds_addr [63:32]
  g0.z = (u32)ga;                                 // global_addr[31:0]
  g0.w = ((u32)(ga >> 32) & 0x01FFFFFFu) | (2u << 30); // ga[56:32] | type=2
  i32x8 g1 = {
    (int)0x00010000u,                                       // data_size=2B
    (int)((tenD0 & 0xFFFFu) << 16),                         // tensor_dim0 lo
    (int)(((tenD0 >> 16) & 0xFFFFu) | ((tenD1 & 0xFFFFu) << 16)),
    (int)(((tenD1 >> 16) & 0xFFFFu) | ((tilD0 & 0xFFFFu) << 16)),
    (int)(tilD1 & 0xFFFFu),                                 // tile_dim1, dim2=0
    (int)strD0,                                             // stride0[31:0]
    0, 0
  };
  i32x4 z4 = {0, 0, 0, 0};
#if __clang_major__ >= 23
  i32x8 z8 = {0, 0, 0, 0, 0, 0, 0, 0};
  __builtin_amdgcn_tensor_load_to_lds(g0, g1, z4, z4, z8, 0);
#else
  __builtin_amdgcn_tensor_load_to_lds(g0, g1, z4, z4, 0);
#endif
}
#endif // USE_TDM

// ---------------------------------------------------------------------------
// fp32 -> bf16 conversion (8 elems/thread) for hidden states
// ---------------------------------------------------------------------------
__global__ void cvt_f32_bf16(const float* __restrict__ in,
                             u16* __restrict__ out, int n8) {
  int i = blockIdx.x * blockDim.x + threadIdx.x;
  if (i >= n8) return;
  const f32x4* p = (const f32x4*)in + (size_t)i * 2;
  f32x4 a = p[0], b = p[1];
  u32x4 o;
  o.x = (u32)f2bf(a.x) | ((u32)f2bf(a.y) << 16);
  o.y = (u32)f2bf(a.z) | ((u32)f2bf(a.w) << 16);
  o.z = (u32)f2bf(b.x) | ((u32)f2bf(b.y) << 16);
  o.w = (u32)f2bf(b.z) | ((u32)f2bf(b.w) << 16);
  ((u32x4*)out)[i] = o;
}

// ---------------------------------------------------------------------------
// fp32 W[k][n] -> bf16 Wt[n][k]: LDS-tiled transpose+convert (64x64 tiles),
// coalesced fp32 reads, coalesced b128 bf16 writes.
// ---------------------------------------------------------------------------
#define TP 66   // padded LDS pitch (u16) to spread banks
__global__ __launch_bounds__(256)
void cvt_transpose_f32_bf16(const float* __restrict__ W, u16* __restrict__ Wt) {
  __shared__ u16 T[64 * TP];
  const int tid = threadIdx.x;
  const int n0 = blockIdx.x * 64, k0 = blockIdx.y * 64;

  const int r0 = tid >> 4, c4 = tid & 15;       // load coords
#pragma unroll
  for (int i = 0; i < 4; ++i) {
    const int row = r0 + i * 16;                // k within tile
    f32x4 v = *(const f32x4*)(W + (size_t)(k0 + row) * EE + n0 + c4 * 4);
    T[(c4 * 4 + 0) * TP + row] = f2bf(v.x);
    T[(c4 * 4 + 1) * TP + row] = f2bf(v.y);
    T[(c4 * 4 + 2) * TP + row] = f2bf(v.z);
    T[(c4 * 4 + 3) * TP + row] = f2bf(v.w);
  }
  __syncthreads();

  const int n = tid >> 2, kc = (tid & 3) << 4;  // store coords
  FragU f;
#pragma unroll
  for (int j = 0; j < 16; ++j) f.s[j] = T[n * TP + kc + j];
  u32x4* dst = (u32x4*)(Wt + (size_t)(n0 + n) * EE + k0 + kc);
  dst[0] = f.q[0]; dst[1] = f.q[1];
}

__global__ void init_min(u32* p) { *p = 0xFFFFFFFFu; }

// ---------------------------------------------------------------------------
// bf16 WMMA GEMM: C[M,2048] = A[M,2048] * Wt^T, Wt pre-transposed [n][k].
// 128x128 block tile, 8 waves x (2x4) 16x16x32 WMMAs, K-step 32,
// TDM double-buffered LDS staging (wave 0 issues, s_wait_tensorcnt, barrier).
// MODE 0: bf16 [B,H,S,D]   MODE 1: fp32 row-major + bias   MODE 2: bf16 [B,H,D,S]
// ---------------------------------------------------------------------------
template <int MODE>
__global__ __launch_bounds__(256)
void gemm_bf16_k2048(const u16* __restrict__ A, const u16* __restrict__ Wt,
                     const float* __restrict__ bias,
                     u16* __restrict__ outB, float* __restrict__ outF) {
  __shared__ u16 As[2][128 * 32];   // [m][k]
  __shared__ u16 Bs[2][128 * 32];   // [n][k]

  const int tid    = threadIdx.x;
  const int lane   = tid & 31;
  const int wave   = tid >> 5;
  const int laneHi = (lane >> 4) & 1;
  const int ln     = lane & 15;
  const int wm     = wave >> 1, wn = wave & 1;
  const int nBase  = blockIdx.x * 128;
  const int mBase  = blockIdx.y * 128;

  const u16* Atile = A  + (size_t)mBase * 2048;
  const u16* Btile = Wt + (size_t)nBase * 2048;

#if !USE_TDM
  const int crow = tid >> 1, ccol = (tid & 1) << 4;   // manual 128x32 copy
#endif

  v8f acc[2][4];
#pragma unroll
  for (int i = 0; i < 2; ++i)
#pragma unroll
    for (int j = 0; j < 4; ++j) acc[i][j] = v8f_zero();

  // prologue: stage k0=0 into buffer 0
#if USE_TDM
  if (wave == 0) {
    tdm_load_2d(lds_raw(&As[0][0]), Atile, 2048, 4096, 32, 128, 2048);
    tdm_load_2d(lds_raw(&Bs[0][0]), Btile, 2048, 2048, 32, 128, 2048);
    __builtin_amdgcn_s_wait_tensorcnt(0);
  }
#else
  {
    const u16* sa = Atile + (size_t)crow * 2048 + ccol;
    const u16* sb = Btile + (size_t)crow * 2048 + ccol;
    u32x4* da = (u32x4*)(&As[0][0] + crow * 32 + ccol);
    u32x4* db = (u32x4*)(&Bs[0][0] + crow * 32 + ccol);
    da[0] = ((const u32x4*)sa)[0]; da[1] = ((const u32x4*)sa)[1];
    db[0] = ((const u32x4*)sb)[0]; db[1] = ((const u32x4*)sb)[1];
  }
#endif
  __syncthreads();

  int cur = 0;
  for (int k0 = 0; k0 < 2048; k0 += 32) {
    const int nxt = cur ^ 1;
    // issue next tile while computing current
#if USE_TDM
    if (wave == 0 && k0 + 32 < 2048) {
      tdm_load_2d(lds_raw(&As[nxt][0]), Atile + k0 + 32, 2048, 4096, 32, 128, 2048);
      tdm_load_2d(lds_raw(&Bs[nxt][0]), Btile + k0 + 32, 2048, 2048, 32, 128, 2048);
    }
#else
    if (k0 + 32 < 2048) {
      const u16* sa = Atile + (size_t)crow * 2048 + k0 + 32 + ccol;
      const u16* sb = Btile + (size_t)crow * 2048 + k0 + 32 + ccol;
      u32x4* da = (u32x4*)(&As[nxt][0] + crow * 32 + ccol);
      u32x4* db = (u32x4*)(&Bs[nxt][0] + crow * 32 + ccol);
      da[0] = ((const u32x4*)sa)[0]; da[1] = ((const u32x4*)sa)[1];
      db[0] = ((const u32x4*)sb)[0]; db[1] = ((const u32x4*)sb)[1];
    }
#endif
    v16bf af[2], bfr[4];
#pragma unroll
    for (int mi = 0; mi < 2; ++mi) {
      const u16* p = &As[cur][0] + (wm * 32 + mi * 16 + ln) * 32 + (laneHi ? 8 : 0);
      af[mi] = make_frag(*(const u32x4*)p, *(const u32x4*)(p + 16));
    }
#pragma unroll
    for (int ni = 0; ni < 4; ++ni) {
      const u16* p = &Bs[cur][0] + (wn * 64 + ni * 16 + ln) * 32 + (laneHi ? 8 : 0);
      bfr[ni] = make_frag(*(const u32x4*)p, *(const u32x4*)(p + 16));
    }
#pragma unroll
    for (int mi = 0; mi < 2; ++mi)
#pragma unroll
      for (int ni = 0; ni < 4; ++ni)
        acc[mi][ni] = wmma_bf16(af[mi], bfr[ni], acc[mi][ni]);

#if USE_TDM
    if (wave == 0) __builtin_amdgcn_s_wait_tensorcnt(0);
#endif
    __syncthreads();
    cur = nxt;
  }

#pragma unroll
  for (int mi = 0; mi < 2; ++mi)
#pragma unroll
    for (int ni = 0; ni < 4; ++ni) {
      const int n = nBase + wn * 64 + ni * 16 + ln;
#pragma unroll
      for (int r = 0; r < 8; ++r) {
        const int m = mBase + wm * 32 + mi * 16 + r + (laneHi ? 8 : 0);
        const float val = acc[mi][ni][r];
        const int b = m >> 11, s = m & 2047, h = n >> 7, d = n & 127;
        if (MODE == 0) {        // [B,H,S,D] bf16 (Q, K)
          outB[((size_t)((b * HH + h) * SS + s)) * DD + d] = f2bf(val);
        } else if (MODE == 2) { // [B,H,D,S] bf16 (V transposed for flash)
          outB[((size_t)((b * HH + h) * DD + d)) * SS + s] = f2bf(val);
        } else {                // fp32 row-major + bias
          outF[(size_t)m * 2048 + n] = val + bias[n];
        }
      }
    }
}

// ---------------------------------------------------------------------------
// Pass A: global min over QK^T (full tensor, pre-mask, per the reference).
// ---------------------------------------------------------------------------
__global__ __launch_bounds__(256)
void attn_min_pass(const u16* __restrict__ Q, const u16* __restrict__ K,
                   u32* __restrict__ minEnc) {
  const int tid = threadIdx.x, lane = tid & 31, wave = tid >> 5;
  const int laneHi = (lane >> 4) & 1, ln = lane & 15;
  const int bh = blockIdx.y;
  const int q0 = blockIdx.x * 128 + wave * 16;
  const u16* Qh = Q + (size_t)bh * SS * DD;
  const u16* Kh = K + (size_t)bh * SS * DD;

  v16bf aq[4];
  {
    const u16* qrow = Qh + (size_t)(q0 + ln) * DD;
#pragma unroll
    for (int d = 0; d < 4; ++d) {
      const int kb = d * 32 + (laneHi ? 8 : 0);
      aq[d] = make_frag(*(const u32x4*)(qrow + kb), *(const u32x4*)(qrow + kb + 16));
    }
  }

  float vmin = 3.0e38f;
  for (int key0 = 0; key0 < SS; key0 += 16) {
    const u16* krow = Kh + (size_t)(key0 + ln) * DD;
    v8f c = v8f_zero();
#pragma unroll
    for (int d = 0; d < 4; ++d) {
      const int kb = d * 32 + (laneHi ? 8 : 0);
      v16bf bk = make_frag(*(const u32x4*)(krow + kb), *(const u32x4*)(krow + kb + 16));
      c = wmma_bf16(aq[d], bk, c);
    }
#pragma unroll
    for (int e = 0; e < 8; ++e) vmin = fminf(vmin, c[e]);
  }
#pragma unroll
  for (int m = 16; m >= 1; m >>= 1) vmin = fminf(vmin, __shfl_xor(vmin, m, 32));
  __shared__ float wmin[8];
  if (lane == 0) wmin[wave] = vmin;
  __syncthreads();
  if (tid == 0) {
    float v = wmin[0];
#pragma unroll
    for (int i = 1; i < 8; ++i) v = fminf(v, wmin[i]);
    const u32 u = __float_as_uint(v);
    const u32 enc = (u & 0x80000000u) ? ~u : (u | 0x80000000u);
    atomicMin(minEnc, enc);
  }
}

// ---------------------------------------------------------------------------
// Pass B: flash attention with finite mask value mv; all keys contribute
// (masked -> mv) to softmax AND P·V, per the reference. V tiles arrive via
// TDM double buffering from pre-transposed V [bh][d][s].
// ---------------------------------------------------------------------------
__global__ __launch_bounds__(256)
void attn_flash(const u16* __restrict__ Q, const u16* __restrict__ K,
                const u16* __restrict__ Vt, const u32* __restrict__ minEnc,
                u16* __restrict__ attnOut) {
  __shared__ u16 VtS[2][128 * 32];  // [d][kk], double-buffered
  __shared__ u16 Pst[8][16 * 32];   // per-wave P staging [q][kk]

  const int tid = threadIdx.x, lane = tid & 31, wave = tid >> 5;
  const int laneHi = (lane >> 4) & 1, ln = lane & 15;
  const int bh = blockIdx.y;
  const int bb = bh >> 4, hh = bh & 15;
  const int q0 = blockIdx.x * 128 + wave * 16;
  const u16* Qh = Q  + (size_t)bh * SS * DD;
  const u16* Kh = K  + (size_t)bh * SS * DD;
  const u16* Vh = Vt + (size_t)bh * SS * DD;   // [d][s]

  const u32 enc = *minEnc;
  const u32 du = (enc & 0x80000000u) ? (enc & 0x7FFFFFFFu) : ~enc;
  const float mv = __uint_as_float(du);

  v16bf aq[4];
  {
    const u16* qrow = Qh + (size_t)(q0 + ln) * DD;
#pragma unroll
    for (int d = 0; d < 4; ++d) {
      const int kb = d * 32 + (laneHi ? 8 : 0);
      aq[d] = make_frag(*(const u32x4*)(qrow + kb), *(const u32x4*)(qrow + kb + 16));
    }
  }

  float mrow[8], lrow[8];
#pragma unroll
  for (int e = 0; e < 8; ++e) { mrow[e] = -3.0e38f; lrow[e] = 0.f; }
  v8f oacc[8];
#pragma unroll
  for (int dt = 0; dt < 8; ++dt) oacc[dt] = v8f_zero();

#if !USE_TDM
  const int vd = tid >> 1, vkc = (tid & 1) << 4;   // manual 128x32 copy coords
#endif

  // prologue: stage V tile key0=0 into buffer 0
#if USE_TDM
  if (wave == 0) {
    tdm_load_2d(lds_raw(&VtS[0][0]), Vh, 2048, 128, 32, 128, 2048);
    __builtin_amdgcn_s_wait_tensorcnt(0);
  }
#else
  {
    const u16* src = Vh + (size_t)vd * SS + vkc;
    u32x4* dst = (u32x4*)(&VtS[0][0] + vd * 32 + vkc);
    dst[0] = ((const u32x4*)src)[0]; dst[1] = ((const u32x4*)src)[1];
  }
#endif
  __syncthreads();

  int cur = 0;
  for (int key0 = 0; key0 < SS; key0 += 32) {
    const int nxt = cur ^ 1;
#if USE_TDM
    if (wave == 0 && key0 + 32 < SS)
      tdm_load_2d(lds_raw(&VtS[nxt][0]), Vh + key0 + 32, 2048, 128, 32, 128, 2048);
#else
    if (key0 + 32 < SS) {
      const u16* src = Vh + (size_t)vd * SS + key0 + 32 + vkc;
      u32x4* dst = (u32x4*)(&VtS[nxt][0] + vd * 32 + vkc);
      dst[0] = ((const u32x4*)src)[0]; dst[1] = ((const u32x4*)src)[1];
    }
#endif

    // scores: two 16(q) x 16(k) tiles covering 32 keys
    v8f c0 = v8f_zero(), c1 = v8f_zero();
    const u16* kr0 = Kh + (size_t)(key0 + ln) * DD;
    const u16* kr1 = Kh + (size_t)(key0 + 16 + ln) * DD;
#pragma unroll
    for (int d = 0; d < 4; ++d) {
      const int kb = d * 32 + (laneHi ? 8 : 0);
      v16bf b0 = make_frag(*(const u32x4*)(kr0 + kb), *(const u32x4*)(kr0 + kb + 16));
      v16bf b1 = make_frag(*(const u32x4*)(kr1 + kb), *(const u32x4*)(kr1 + kb + 16));
      c0 = wmma_bf16(aq[d], b0, c0);
      c1 = wmma_bf16(aq[d], b1, c1);
    }

    // mask -> mv, online softmax update, build bf16 P in per-wave LDS
#pragma unroll
    for (int e = 0; e < 8; ++e) {
      const int qrow = q0 + e + (laneHi ? 8 : 0);
      const float s0 = ((key0 + ln) > qrow) ? mv : c0[e];
      const float s1 = ((key0 + 16 + ln) > qrow) ? mv : c1[e];
      float tmax = fmaxf(s0, s1);
#pragma unroll
      for (int m = 8; m >= 1; m >>= 1) tmax = fmaxf(tmax, __shfl_xor(tmax, m, 32));
      const float mnew  = fmaxf(mrow[e], tmax);
      const float scale = __expf(mrow[e] - mnew);
      const float e0 = __expf(s0 - mnew);
      const float e1 = __expf(s1 - mnew);
      float tsum = e0 + e1;
#pragma unroll
      for (int m = 8; m >= 1; m >>= 1) tsum += __shfl_xor(tsum, m, 32);
      lrow[e] = lrow[e] * scale + tsum;
      mrow[e] = mnew;
#pragma unroll
      for (int dt = 0; dt < 8; ++dt) oacc[dt][e] *= scale;
      Pst[wave][(e + (laneHi ? 8 : 0)) * 32 + ln]      = f2bf(e0);
      Pst[wave][(e + (laneHi ? 8 : 0)) * 32 + 16 + ln] = f2bf(e1);
    }

    { // P (A-layout via wave-private LDS) x V (B-layout from VtS[cur])
      const u16* pp = &Pst[wave][0] + ln * 32 + (laneHi ? 8 : 0);
      v16bf ap = make_frag(*(const u32x4*)pp, *(const u32x4*)(pp + 16));
#pragma unroll
      for (int dt = 0; dt < 8; ++dt) {
        const u16* vp = &VtS[cur][0] + (dt * 16 + ln) * 32 + (laneHi ? 8 : 0);
        v16bf bv = make_frag(*(const u32x4*)vp, *(const u32x4*)(vp + 16));
        oacc[dt] = wmma_bf16(ap, bv, oacc[dt]);
      }
    }
#if USE_TDM
    if (wave == 0) __builtin_amdgcn_s_wait_tensorcnt(0);
#endif
    __syncthreads();
    cur = nxt;
  }

  // normalize and write merged-heads bf16 [B,S,E]
#pragma unroll
  for (int dt = 0; dt < 8; ++dt)
#pragma unroll
    for (int e = 0; e < 8; ++e) {
      const int qrow = q0 + e + (laneHi ? 8 : 0);
      const int col  = hh * DD + dt * 16 + ln;
      attnOut[(size_t)(bb * SS + qrow) * EE + col] = f2bf(oacc[dt][e] / lrow[e]);
    }
}

// ---------------------------------------------------------------------------
// Host-side launcher
// ---------------------------------------------------------------------------
extern "C" void kernel_launch(void* const* d_in, const int* in_sizes, int n_in,
                              void* d_out, int out_size, void* d_ws, size_t ws_size,
                              hipStream_t stream) {
  const float* hs = (const float*)d_in[0];
  const float* wq = (const float*)d_in[1];
  const float* wk = (const float*)d_in[2];
  const float* wv = (const float*)d_in[3];
  const float* wo = (const float*)d_in[4];
  const float* bo = (const float*)d_in[5];
  float* out = (float*)d_out;

  char* ws = (char*)d_ws;
  u16* Xb  = (u16*)(ws + 0);            // 4096x2048 bf16 X
  u16* Wqb = (u16*)(ws + 16777216);     // 2048x2048 bf16, TRANSPOSED [n][k]
  u16* Wkb = (u16*)(ws + 25165824);
  u16* Wvb = (u16*)(ws + 33554432);
  u16* Wob = (u16*)(ws + 41943040);
  u16* Qh  = (u16*)(ws + 50331648);     // [B,H,S,D] bf16
  u16* Kh  = (u16*)(ws + 67108864);     // [B,H,S,D] bf16
  u16* Vh  = (u16*)(ws + 83886080);     // [B,H,D,S] bf16 (transposed)
  u16* Ab  = (u16*)(ws + 100663296);    // merged heads [B,S,E] bf16
  u32* minEnc = (u32*)(ws + 117440512);

  cvt_f32_bf16<<<(8388608 / 8) / 256, 256, 0, stream>>>(hs, Xb, 8388608 / 8);
  const dim3 gT(32, 32);                // 64x64 transpose tiles
  cvt_transpose_f32_bf16<<<gT, 256, 0, stream>>>(wq, Wqb);
  cvt_transpose_f32_bf16<<<gT, 256, 0, stream>>>(wk, Wkb);
  cvt_transpose_f32_bf16<<<gT, 256, 0, stream>>>(wv, Wvb);
  cvt_transpose_f32_bf16<<<gT, 256, 0, stream>>>(wo, Wob);
  init_min<<<1, 1, 0, stream>>>(minEnc);

  const dim3 gProj(16, 32), blk(256);   // N/128=16, M/128=32
  gemm_bf16_k2048<0><<<gProj, blk, 0, stream>>>(Xb, Wqb, nullptr, Qh, nullptr);
  gemm_bf16_k2048<0><<<gProj, blk, 0, stream>>>(Xb, Wkb, nullptr, Kh, nullptr);
  gemm_bf16_k2048<2><<<gProj, blk, 0, stream>>>(Xb, Wvb, nullptr, Vh, nullptr);

  const dim3 gAttn(16, 32);             // q-tiles of 128, B*H=32
  attn_min_pass<<<gAttn, blk, 0, stream>>>(Qh, Kh, minEnc);
  attn_flash<<<gAttn, blk, 0, stream>>>(Qh, Kh, Vh, minEnc, Ab);

  gemm_bf16_k2048<1><<<gProj, blk, 0, stream>>>(Ab, Wob, bo, nullptr, out);
}
